// DynamicPointSegmentor_20014547599826
// MI455X (gfx1250) — matrix operand order repaired and checked by reading, not compiled, not run on hardware
//
#include <hip/hip_runtime.h>

#define N_PTS 250000
#define N_VOX 60000
#define D_PT 256
#define D_VOX 256
#define MID 256
#define NUM_CLASS 20
#define KDIM 512
#define BN_EPS 1e-5f

typedef __attribute__((ext_vector_type(16))) __bf16 v16bf;
typedef __attribute__((ext_vector_type(8)))  float  v8f;
typedef __attribute__((ext_vector_type(4)))  int    v4i;

#if __has_builtin(__builtin_amdgcn_global_load_async_to_lds_b128) && \
    __has_builtin(__builtin_amdgcn_s_wait_asynccnt)
#define HAVE_ASYNC_LDS 1
typedef __attribute__((address_space(1))) v4i* gas_v4i_ptr;
typedef __attribute__((address_space(3))) v4i* las_v4i_ptr;
#endif

// ---------------------------------------------------------------- utilities
__global__ __launch_bounds__(256) void zero_kernel(float* p, int n) {
    int i = blockIdx.x * 256 + threadIdx.x;
    if (i < n) p[i] = 0.0f;
}

__global__ __launch_bounds__(256) void cvt_w1_kernel(const float* W1, __bf16* w1bf) {
    int i = blockIdx.x * 256 + threadIdx.x;   // MID*KDIM = 131072 elements
    if (i < MID * KDIM) w1bf[i] = (__bf16)W1[i];
}

// ------------------------------------------ softmax + cls_fea = prob^T @ vf
__global__ __launch_bounds__(256) void clsfea_kernel(const float* __restrict__ logits,
                                                     const float* __restrict__ vf,
                                                     float* __restrict__ clsacc) {
    __shared__ float p[256][NUM_CLASS + 1];
    int t = threadIdx.x;
    int vbase = blockIdx.x * 256;
    int v = vbase + t;
    if (v < N_VOX) {
        float l[NUM_CLASS];
        float mx = -1e30f;
        #pragma unroll
        for (int c = 0; c < NUM_CLASS; c++) { l[c] = logits[(size_t)v * NUM_CLASS + c]; mx = fmaxf(mx, l[c]); }
        float s = 0.f;
        #pragma unroll
        for (int c = 0; c < NUM_CLASS; c++) { l[c] = __expf(l[c] - mx); s += l[c]; }
        float inv = 1.f / s;
        #pragma unroll
        for (int c = 0; c < NUM_CLASS; c++) p[t][c] = l[c] * inv;
    }
    __syncthreads();
    int nv = min(256, N_VOX - vbase);
    float acc[NUM_CLASS] = {};
    for (int vv = 0; vv < nv; vv++) {
        float f = vf[(size_t)(vbase + vv) * D_VOX + t];   // coalesced: t = feature index
        #pragma unroll
        for (int c = 0; c < NUM_CLASS; c++) acc[c] += p[vv][c] * f;
    }
    #pragma unroll
    for (int c = 0; c < NUM_CLASS; c++) atomicAdd(&clsacc[c * D_VOX + t], acc[c]);
}

// ------------------------------------ cls2 = cls_fea @ W2^T + b2  (20x256)
__global__ __launch_bounds__(256) void cls2_kernel(const float* __restrict__ clsacc,
                                                   const float* __restrict__ W2,
                                                   const float* __restrict__ b2,
                                                   float* __restrict__ cls2) {
    __shared__ float row[MID];
    int c = blockIdx.x, t = threadIdx.x;
    row[t] = clsacc[c * MID + t];
    __syncthreads();
    float sum = b2[t];
    for (int k = 0; k < MID; k++) sum += row[k] * W2[(size_t)t * MID + k];
    cls2[c * MID + t] = sum;
}

// ------------------------------------------- BN over the 20 class rows
__global__ __launch_bounds__(256) void clsbn_kernel(const float* __restrict__ cls2,
                                                    const float* __restrict__ g2,
                                                    const float* __restrict__ be2,
                                                    float* __restrict__ clsfin) {
    int j = threadIdx.x;
    float m = 0.f;
    for (int c = 0; c < NUM_CLASS; c++) m += cls2[c * MID + j];
    m *= (1.0f / NUM_CLASS);
    float var = 0.f;
    for (int c = 0; c < NUM_CLASS; c++) { float d = cls2[c * MID + j] - m; var += d * d; }
    var *= (1.0f / NUM_CLASS);
    float sc = g2[j] * rsqrtf(var + BN_EPS);
    float sh = be2[j];
    for (int c = 0; c < NUM_CLASS; c++) clsfin[c * MID + j] = (cls2[c * MID + j] - m) * sc + sh;
}

// --------------------- big WMMA GEMM: y1 = [pf | vf[p2v]] @ W1^T + b1
// block = 128 threads (4 waves), tile M=64 N=64, K-step 32, bf16 in / f32 acc.
// A fragments load straight from global (no reuse across lanes); B is staged
// through LDS in fragment-native order via async global->LDS b128 copies.
#define TM 64
#define TN 64
#define TK 32
__global__ __launch_bounds__(128) void gemm1_kernel(const float* __restrict__ pf,
                                                    const float* __restrict__ vf,
                                                    const long long* __restrict__ p2v,
                                                    const __bf16* __restrict__ w1bf,
                                                    const float* __restrict__ b1,
                                                    float* __restrict__ y1,
                                                    float* __restrict__ colsum,
                                                    float* __restrict__ colsumsq) {
    // B tile in fragment order: entry (nn*32+lane) holds that lane's 16 bf16, contiguous
    __shared__ __align__(16) __bf16 Bf[128][16];
    __shared__ float s_cs[TN], s_cs2[TN];

    int t = threadIdx.x;
    int mBase = blockIdx.x * TM;
    int nBase = blockIdx.y * TN;
    int wave = t >> 5, lane = t & 31;

    if (t < TN) { s_cs[t] = 0.f; s_cs2[t] = 0.f; }

    int lrow = lane & 15;                 // A: row within strip / B,C: column within 16
    int akb  = (lane < 16) ? 0 : 16;      // A-matrix K base (contiguous K: k = 16h + s, matches B)
    int mrow = mBase + wave * 16 + lrow;
    int mclamp = min(mrow, N_PTS - 1);    // rows past N_PTS: computed, never stored
    long long vox = p2v[mclamp];
    const float* prow = pf + (size_t)mclamp * D_PT;
    const float* vrow = vf + (size_t)vox * D_VOX;

    // B staging: thread t owns fragment entry t (pure bf16 copy)
    int bq_n  = nBase + (t >> 5) * 16 + (t & 15);        // output column
    int bq_kb = ((t & 31) < 16) ? 0 : 16;                // lane-group K base
    const __bf16* bq_src = w1bf + (size_t)bq_n * KDIM + bq_kb;

    v8f acc[4] = {};

    for (int k0 = 0; k0 < KDIM; k0 += TK) {
        // ---- stage B into LDS (async fire-and-forget) ----
        {
            uint4* gsrc = (uint4*)(bq_src + k0);
            uint4* ldst = (uint4*)&Bf[t][0];
#if defined(HAVE_ASYNC_LDS)
            __builtin_amdgcn_global_load_async_to_lds_b128(
                (gas_v4i_ptr)gsrc, (las_v4i_ptr)ldst, 0, 0);
            __builtin_amdgcn_global_load_async_to_lds_b128(
                (gas_v4i_ptr)gsrc, (las_v4i_ptr)ldst, 16, 0);
#else
            ldst[0] = gsrc[0];
            ldst[1] = gsrc[1];
#endif
        }
        // ---- A fragment straight from global: 4x b128 + 8x cvt_pk_bf16 ----
        const float* arow = (k0 < D_PT) ? (prow + k0) : (vrow + (k0 - D_PT));
        float4 f0 = *(const float4*)(arow + akb);
        float4 f1 = *(const float4*)(arow + akb + 4);
        float4 f2 = *(const float4*)(arow + akb + 8);
        float4 f3 = *(const float4*)(arow + akb + 12);
        union { __bf16 h[16]; v16bf v; } af;
        af.h[0]  = (__bf16)f0.x; af.h[1]  = (__bf16)f0.y;
        af.h[2]  = (__bf16)f0.z; af.h[3]  = (__bf16)f0.w;
        af.h[4]  = (__bf16)f1.x; af.h[5]  = (__bf16)f1.y;
        af.h[6]  = (__bf16)f1.z; af.h[7]  = (__bf16)f1.w;
        af.h[8]  = (__bf16)f2.x; af.h[9]  = (__bf16)f2.y;
        af.h[10] = (__bf16)f2.z; af.h[11] = (__bf16)f2.w;
        af.h[12] = (__bf16)f3.x; af.h[13] = (__bf16)f3.y;
        af.h[14] = (__bf16)f3.z; af.h[15] = (__bf16)f3.w;

#if defined(HAVE_ASYNC_LDS)
        __builtin_amdgcn_s_wait_asynccnt(0);
#endif
        __syncthreads();

        #pragma unroll
        for (int nn = 0; nn < 4; nn++) {
            union { uint4 u[2]; v16bf v; } bf;
            const uint4* bp = (const uint4*)&Bf[nn * 32 + lane][0];
            bf.u[0] = bp[0];
            bf.u[1] = bp[1];
            acc[nn] = __builtin_amdgcn_wmma_f32_16x16x32_bf16(
                false, af.v, false, bf.v, (short)0, acc[nn], false, false);
        }
        __syncthreads();
    }

    // epilogue: bias, store, BN column sums (C/D layout: lane n=l&15, mb=(l<16?0:8))
    int mb = (lane < 16) ? 0 : 8;
    #pragma unroll
    for (int nn = 0; nn < 4; nn++) {
        int ncol = nBase + nn * 16 + lrow;
        float bias = b1[ncol];
        float ps = 0.f, ps2 = 0.f;
        #pragma unroll
        for (int r = 0; r < 8; r++) {
            int m = mBase + wave * 16 + mb + r;
            if (m < N_PTS) {
                float v = acc[nn][r] + bias;
                y1[(size_t)m * MID + ncol] = v;
                ps += v; ps2 += v * v;
            }
        }
        atomicAdd(&s_cs[nn * 16 + lrow], ps);
        atomicAdd(&s_cs2[nn * 16 + lrow], ps2);
    }
    __syncthreads();
    if (t < TN) {
        atomicAdd(&colsum[nBase + t], s_cs[t]);
        atomicAdd(&colsumsq[nBase + t], s_cs2[t]);
    }
}

// -------------------------------- finalize BN affine: scale/shift per column
__global__ __launch_bounds__(256) void bnstats_kernel(const float* __restrict__ colsum,
                                                      const float* __restrict__ colsumsq,
                                                      const float* __restrict__ g1,
                                                      const float* __restrict__ be1,
                                                      float* __restrict__ scale,
                                                      float* __restrict__ shift) {
    int j = threadIdx.x;
    float inv = 1.0f / (float)N_PTS;
    float mean = colsum[j] * inv;
    float var  = colsumsq[j] * inv - mean * mean;
    float s = g1[j] * rsqrtf(var + BN_EPS);
    scale[j] = s;
    shift[j] = be1[j] - mean * s;
}

// ------------- fused BN + ReLU + final GEMM: out = relu(bn(y1)) @ clsfin^T
__global__ __launch_bounds__(256) void final_kernel(const float* __restrict__ y1,
                                                    const float* __restrict__ scale,
                                                    const float* __restrict__ shift,
                                                    const float* __restrict__ clsfin,
                                                    float* __restrict__ out) {
    __shared__ float cls[NUM_CLASS][MID];
    __shared__ float xs[16][MID];
    int t = threadIdx.x;
    for (int i = t; i < NUM_CLASS * MID; i += 256) cls[i / MID][i % MID] = clsfin[i];
    int mBase = blockIdx.x * 16;
    float s = scale[t], sh = shift[t];
    #pragma unroll
    for (int r = 0; r < 16; r++) {
        float v = y1[(size_t)(mBase + r) * MID + t] * s + sh;   // coalesced
        xs[r][t] = v > 0.f ? v : 0.f;
    }
    __syncthreads();
    for (int o = t; o < 16 * NUM_CLASS; o += 256) {
        int r = o / NUM_CLASS, c = o % NUM_CLASS;
        float sum = 0.f;
        for (int j = 0; j < MID; j++) sum += xs[r][j] * cls[c][j];
        out[(size_t)(mBase + r) * NUM_CLASS + c] = sum;
    }
}

// ---------------------------------------------------------------- launcher
extern "C" void kernel_launch(void* const* d_in, const int* in_sizes, int n_in,
                              void* d_out, int out_size, void* d_ws, size_t ws_size,
                              hipStream_t stream) {
    const float* pf     = (const float*)d_in[0];
    const float* vf     = (const float*)d_in[1];
    const float* logits = (const float*)d_in[2];
    const float* W1     = (const float*)d_in[3];
    const float* b1     = (const float*)d_in[4];
    const float* g1     = (const float*)d_in[5];
    const float* be1    = (const float*)d_in[6];
    const float* W2     = (const float*)d_in[7];
    const float* b2     = (const float*)d_in[8];
    const float* g2     = (const float*)d_in[9];
    const float* be2    = (const float*)d_in[10];
    const long long* p2v = (const long long*)d_in[11];

    char* w = (char*)d_ws;
    size_t o = 0;
    float*  y1      = (float*)(w + o);  o += (size_t)N_PTS * MID * sizeof(float);     // 256 MB
    __bf16* w1bf    = (__bf16*)(w + o); o += (size_t)MID * KDIM * sizeof(__bf16);     // 256 KB
    float*  clsacc  = (float*)(w + o);  o += (size_t)NUM_CLASS * MID * sizeof(float); // contiguous with
    float*  colsum  = (float*)(w + o);  o += (size_t)MID * sizeof(float);             // colsum / colsumsq
    float*  colsumsq= (float*)(w + o);  o += (size_t)MID * sizeof(float);             // for one zero pass
    float*  cls2    = (float*)(w + o);  o += (size_t)NUM_CLASS * MID * sizeof(float);
    float*  clsfin  = (float*)(w + o);  o += (size_t)NUM_CLASS * MID * sizeof(float);
    float*  scl     = (float*)(w + o);  o += (size_t)MID * sizeof(float);
    float*  shf     = (float*)(w + o);  o += (size_t)MID * sizeof(float);
    (void)ws_size; (void)in_sizes; (void)n_in; (void)out_size;

    int nzero = NUM_CLASS * MID + 2 * MID;   // clsacc + colsum + colsumsq
    zero_kernel<<<(nzero + 255) / 256, 256, 0, stream>>>(clsacc, nzero);
    cvt_w1_kernel<<<(MID * KDIM + 255) / 256, 256, 0, stream>>>(W1, w1bf);
    clsfea_kernel<<<(N_VOX + 255) / 256, 256, 0, stream>>>(logits, vf, clsacc);
    cls2_kernel<<<NUM_CLASS, 256, 0, stream>>>(clsacc, W2, b2, cls2);
    clsbn_kernel<<<1, 256, 0, stream>>>(cls2, g2, be2, clsfin);
    dim3 g1grid((N_PTS + TM - 1) / TM, MID / TN);
    gemm1_kernel<<<g1grid, 128, 0, stream>>>(pf, vf, p2v, w1bf, b1, y1, colsum, colsumsq);
    bnstats_kernel<<<1, 256, 0, stream>>>(colsum, colsumsq, g1, be1, scl, shf);
    final_kernel<<<N_PTS / 16, 256, 0, stream>>>(y1, scl, shf, clsfin, (float*)d_out);
}
